// LlamaAttention_24696061952347
// MI455X (gfx1250) — matrix operand order repaired and checked by reading, not compiled, not run on hardware
//
#include <hip/hip_runtime.h>
#include <stdint.h>

typedef __bf16 bf16_t;
typedef __attribute__((ext_vector_type(16))) __bf16 v16bf;
typedef __attribute__((ext_vector_type(8)))  float  v8f;

#define S_LEN 2048
#define HID   4096
#define NH    32
#define NKV   8
#define HD    128
#define KVD   (NKV * HD)   // 1024

// ---------- helpers ----------

// Build a 16-element bf16 fragment from two 16-byte chunks (global or LDS).
static __device__ __forceinline__ v16bf load_frag16(const bf16_t* p0, const bf16_t* p1) {
    union { uint4 q[2]; v16bf v; } u;
    u.q[0] = *reinterpret_cast<const uint4*>(p0);
    u.q[1] = *reinterpret_cast<const uint4*>(p1);
    return u.v;
}

static __device__ __forceinline__ v8f wmma_bf16(v16bf a, v16bf b, v8f c) {
    // D = A(16x32) * B(32x16) + C, fp32 accumulate
    return __builtin_amdgcn_wmma_f32_16x16x32_bf16(
        /*neg_a=*/false, a, /*neg_b=*/false, b,
        /*c_mod=*/(short)0, c, /*reuse_a=*/false, /*reuse_b=*/false);
}

// LDS byte offset of a __shared__ object: generic LDS pointers carry the
// aperture in bits [63:32]; bits [31:0] are the in-LDS offset.
static __device__ __forceinline__ uint32_t lds_offset(const void* p) {
    return (uint32_t)(uintptr_t)p;
}

// CDNA5 async global -> LDS copy (ASYNCcnt-tracked, bypasses VGPRs).
static __device__ __forceinline__ void async_load_b128(uint32_t lds_off, const bf16_t* g) {
    asm volatile("global_load_async_to_lds_b128 %0, %1, off"
                 :: "v"(lds_off), "v"((unsigned long long)(uintptr_t)g)
                 : "memory");
}

static __device__ __forceinline__ void wait_async0() {
    asm volatile("s_wait_asynccnt 0" ::: "memory");
}

// ---- DPP cross-lane exchange (VALU-speed, no LDS pipe) ----
// lane-select patterns for xor within 8-lane groups
#define DPP8_SEL(a,b,c,d,e,f,g,h) \
    ((a)|((b)<<3)|((c)<<6)|((d)<<9)|((e)<<12)|((f)<<15)|((g)<<18)|((h)<<21))
constexpr int kDpp8Xor1 = DPP8_SEL(1,0,3,2,5,4,7,6);
constexpr int kDpp8Xor2 = DPP8_SEL(2,3,0,1,6,7,4,5);
constexpr int kDpp8Xor4 = DPP8_SEL(4,5,6,7,0,1,2,3);

template<int SEL>
static __device__ __forceinline__ float dpp8_xchg(float x) {
    return __int_as_float(__builtin_amdgcn_mov_dpp8(__float_as_int(x), SEL));
}
// xor 8 within a 16-lane DPP row == row_ror:8 (ctrl 0x128)
static __device__ __forceinline__ float dpp_row_ror8(float x) {
    int i = __float_as_int(x);
    i = __builtin_amdgcn_update_dpp(i, i, 0x128, 0xF, 0xF, true);
    return __int_as_float(i);
}
// reduce across the 16-lane half that owns a C-fragment row
static __device__ __forceinline__ float half_reduce_max(float v) {
    v = fmaxf(v, dpp8_xchg<kDpp8Xor1>(v));
    v = fmaxf(v, dpp8_xchg<kDpp8Xor2>(v));
    v = fmaxf(v, dpp8_xchg<kDpp8Xor4>(v));
    v = fmaxf(v, dpp_row_ror8(v));
    return v;
}
static __device__ __forceinline__ float half_reduce_sum(float v) {
    v += dpp8_xchg<kDpp8Xor1>(v);
    v += dpp8_xchg<kDpp8Xor2>(v);
    v += dpp8_xchg<kDpp8Xor4>(v);
    v += dpp_row_ror8(v);
    return v;
}

// ---------- fp32 -> bf16 conversion (8 elements / thread / step) ----------

__global__ __launch_bounds__(256)
void cvt_f32_bf16_kernel(const float* __restrict__ src, bf16_t* __restrict__ dst, size_t n8) {
    size_t i = (size_t)blockIdx.x * blockDim.x + threadIdx.x;
    size_t stride = (size_t)gridDim.x * blockDim.x;
    for (; i < n8; i += stride) {
        const float4* s = reinterpret_cast<const float4*>(src + i * 8);
        float4 a = s[0], b = s[1];
        union { uint4 q; bf16_t e[8]; } u;
        u.e[0] = (bf16_t)a.x; u.e[1] = (bf16_t)a.y; u.e[2] = (bf16_t)a.z; u.e[3] = (bf16_t)a.w;
        u.e[4] = (bf16_t)b.x; u.e[5] = (bf16_t)b.y; u.e[6] = (bf16_t)b.z; u.e[7] = (bf16_t)b.w;
        *reinterpret_cast<uint4*>(dst + i * 8) = u.q;
    }
}

// ---------- NT GEMM: C[M,N] = A[M,K] * B[N,K]^T  (A,B bf16 row-major) ----------
// Block tile 128(M) x 256(N), K-step 32, 8 waves as 2x4 of 64x64 wave tiles.
// LDS double-buffered, filled with global_load_async_to_lds_b128.
// OUT_MODE: 0 = fp32 row-major, 2 = bf16 transposed (C^T[N,M])
template<int OUT_MODE>
__global__ __launch_bounds__(256)
void gemm_nt_wmma_kernel(const bf16_t* __restrict__ A,
                         const bf16_t* __restrict__ Bw,
                         void* __restrict__ Cout,
                         int M, int N, int K) {
    // Rows padded to 40 bf16 (80B) -> fragment b128 reads are bank-conflict-free.
    __shared__ bf16_t As[2][128 * 40];
    __shared__ bf16_t Bs[2][256 * 40];

    const int tid  = threadIdx.x;
    const int lane = tid & 31;
    const int wid  = tid >> 5;       // 8 waves
    const int half = lane >> 4;
    const int r    = lane & 15;
    const int mBlk = blockIdx.y * 128;
    const int nBlk = blockIdx.x * 256;
    const int wm   = wid >> 2;       // 0..1 : 64 rows
    const int wn   = wid & 3;        // 0..3 : 64 cols

    v8f acc[4][4];
#pragma unroll
    for (int i = 0; i < 4; ++i)
#pragma unroll
        for (int j = 0; j < 4; ++j)
#pragma unroll
            for (int e = 0; e < 8; ++e) acc[i][j][e] = 0.0f;

    // Cooperative staging assignment:
    //   A tile 128x32: thread t -> row t/2, 16 cols at (t&1)*16  (2 x b128)
    //   B tile 256x32: thread t -> row t,   32 cols              (4 x b128)
    const int lrowA = tid >> 1;
    const int lcolA = (tid & 1) * 16;
    const int lrowB = tid;
    const bf16_t* gA = A  + (size_t)(mBlk + lrowA) * K + lcolA;
    const bf16_t* gB = Bw + (size_t)(nBlk + lrowB) * K;

    uint32_t ldsA[2], ldsB[2];
#pragma unroll
    for (int b = 0; b < 2; ++b) {
        ldsA[b] = lds_offset(&As[b][lrowA * 40 + lcolA]);
        ldsB[b] = lds_offset(&Bs[b][lrowB * 40]);
    }

    const int koffA = half * 8;      // A frag: K {koffA..+7, 16+koffA..+7}
    const int koffB = half * 8;      // B frag: K {koffB..+7, 16+koffB..+7}

    // prologue: stage first K-tile into buffer 0
    {
        async_load_b128(ldsA[0],      gA);
        async_load_b128(ldsA[0] + 16, gA + 8);
        async_load_b128(ldsB[0],      gB);
        async_load_b128(ldsB[0] + 16, gB + 8);
        async_load_b128(ldsB[0] + 32, gB + 16);
        async_load_b128(ldsB[0] + 48, gB + 24);
    }

    for (int kc = 0, it = 0; kc < K; kc += 32, ++it) {
        const int cur = it & 1;
        const int nxt = cur ^ 1;

        // current tile's async fills complete + all waves past previous compute
        wait_async0();
        __syncthreads();

        // prefetch next K-tile while we compute on `cur`
        if (kc + 32 < K) {
            const int kn = kc + 32;
            async_load_b128(ldsA[nxt],      gA + kn);
            async_load_b128(ldsA[nxt] + 16, gA + kn + 8);
            async_load_b128(ldsB[nxt],      gB + kn);
            async_load_b128(ldsB[nxt] + 16, gB + kn + 8);
            async_load_b128(ldsB[nxt] + 32, gB + kn + 16);
            async_load_b128(ldsB[nxt] + 48, gB + kn + 24);
        }

        v16bf af[4], bfm[4];
#pragma unroll
        for (int i = 0; i < 4; ++i) {
            const int rr = wm * 64 + i * 16 + r;
            af[i] = load_frag16(&As[cur][rr * 40 + koffA], &As[cur][rr * 40 + 16 + koffA]);
        }
#pragma unroll
        for (int j = 0; j < 4; ++j) {
            const int rr = wn * 64 + j * 16 + r;
            bfm[j] = load_frag16(&Bs[cur][rr * 40 + koffB], &Bs[cur][rr * 40 + 16 + koffB]);
        }
#pragma unroll
        for (int i = 0; i < 4; ++i)
#pragma unroll
            for (int j = 0; j < 4; ++j)
                acc[i][j] = wmma_bf16(af[i], bfm[j], acc[i][j]);
    }

    // Epilogue. C layout: lane<16 holds N=r rows M 0..7; lane>=16 rows 8..15.
#pragma unroll
    for (int i = 0; i < 4; ++i) {
#pragma unroll
        for (int j = 0; j < 4; ++j) {
            const int row0 = mBlk + wm * 64 + i * 16 + half * 8;
            const int col  = nBlk + wn * 64 + j * 16 + r;
            if (OUT_MODE == 0) {
#pragma unroll
                for (int e = 0; e < 8; ++e)
                    ((float*)Cout)[(size_t)(row0 + e) * N + col] = acc[i][j][e];
            } else {
                // 8 consecutive rows of C^T[N,M] -> one 16B store
                union { uint4 q; bf16_t e[8]; } u;
#pragma unroll
                for (int e = 0; e < 8; ++e) u.e[e] = (bf16_t)acc[i][j][e];
                *reinterpret_cast<uint4*>(&((bf16_t*)Cout)[(size_t)col * M + row0]) = u.q;
            }
        }
    }
}

// ---------- RoPE: rotate fp32 [S, nHeads*128] and emit bf16 ----------

__global__ __launch_bounds__(128)
void rope_kernel(const float* __restrict__ src, bf16_t* __restrict__ dst, int nHeads) {
    const int s = blockIdx.x;
    const int h = blockIdx.y;
    const int d = threadIdx.x;           // 0..127
    const size_t base = (size_t)s * ((size_t)nHeads * HD) + (size_t)h * HD;
    const float x = src[base + d];
    const int d2 = d & 63;
    // inv_freq = theta^(-2*d2/128) = exp(-d2 * ln(10000)/64)
    const float freq = __expf(-(float)d2 * (9.210340371976184f / 64.0f));
    const float ang = (float)s * freq;
    const float c = __cosf(ang), sn = __sinf(ang);
    const float partner = src[base + ((d < 64) ? (d + 64) : (d - 64))];
    const float rot = (d < 64) ? -partner : partner;
    dst[base + d] = (bf16_t)(x * c + rot * sn);
}

// ---------- Flash attention: 1 wave handles 16 query rows of one head ----------
// Q bf16 [S, NH*128]; K bf16 [S, NKV*128]; VT bf16 [NKV*128, S]; out bf16 [S, NH*128]

__global__ __launch_bounds__(32)
void flash_attn_wmma_kernel(const bf16_t* __restrict__ Q,
                            const bf16_t* __restrict__ Kb,
                            const bf16_t* __restrict__ VT,
                            bf16_t* __restrict__ Oattn) {
    __shared__ bf16_t pS[16 * 40];   // P tile staging (16 x 32 bf16, padded)

    const int lane = threadIdx.x & 31;
    const int half = lane >> 4;
    const int r    = lane & 15;
    const int qBase = blockIdx.x * 16;
    const int h     = blockIdx.y;
    const int kvh   = h >> 2;               // GQA: 4 query heads per KV head
    const float scale = 0.08838834764831845f;   // 1/sqrt(128)

    // Preload Q A-fragments: 4 chunks of K(=d)=32
    v16bf qf[4];
    {
        const bf16_t* qrow = Q + (size_t)(qBase + r) * HID + (size_t)h * HD;
        const int koff = half * 8;
#pragma unroll
        for (int c = 0; c < 4; ++c)
            qf[c] = load_frag16(qrow + c * 32 + koff, qrow + c * 32 + 16 + koff);
    }

    v8f o[8];
    float rm[8], rl[8];
#pragma unroll
    for (int t = 0; t < 8; ++t)
#pragma unroll
        for (int e = 0; e < 8; ++e) o[t][e] = 0.0f;
#pragma unroll
    for (int e = 0; e < 8; ++e) { rm[e] = -3.0e38f; rl[e] = 0.0f; }

    const int kEnd = qBase + 16;   // keys <= qBase+15

    for (int kb = 0; kb < kEnd; kb += 32) {
        // ---- S = Q * K^T for 32 keys (two 16-col tiles) ----
        v8f s0, s1;
#pragma unroll
        for (int e = 0; e < 8; ++e) { s0[e] = 0.0f; s1[e] = 0.0f; }
#pragma unroll
        for (int c = 0; c < 4; ++c) {
            const bf16_t* k0 = Kb + (size_t)(kb + r)      * KVD + (size_t)kvh * HD + c * 32 + half * 8;
            const bf16_t* k1 = Kb + (size_t)(kb + 16 + r) * KVD + (size_t)kvh * HD + c * 32 + half * 8;
            s0 = wmma_bf16(qf[c], load_frag16(k0, k0 + 16), s0);
            s1 = wmma_bf16(qf[c], load_frag16(k1, k1 + 16), s1);
        }

        // scale + causal mask (element e: query = qBase+half*8+e, key col = r / r+16)
        const bool needMask = (kb + 31 > qBase);
#pragma unroll
        for (int e = 0; e < 8; ++e) {
            const int q_idx = qBase + half * 8 + e;
            s0[e] *= scale;
            s1[e] *= scale;
            if (needMask) {
                if (kb + r      > q_idx) s0[e] = -3.0e38f;
                if (kb + 16 + r > q_idx) s1[e] = -3.0e38f;
            }
        }

        // online softmax (row stats via DPP reduce across the owning 16 lanes)
        float sf[8];
#pragma unroll
        for (int e = 0; e < 8; ++e) {
            const float v = half_reduce_max(fmaxf(s0[e], s1[e]));
            const float mn = fmaxf(rm[e], v);
            sf[e] = __expf(rm[e] - mn);
            rm[e] = mn;
        }
#pragma unroll
        for (int e = 0; e < 8; ++e) {
            s0[e] = __expf(s0[e] - rm[e]);
            s1[e] = __expf(s1[e] - rm[e]);
            rl[e] = rl[e] * sf[e] + half_reduce_sum(s0[e] + s1[e]);
        }
#pragma unroll
        for (int t = 0; t < 8; ++t)
#pragma unroll
            for (int e = 0; e < 8; ++e) o[t][e] *= sf[e];

        // ---- stage P (16x32) to LDS, reload in A-fragment layout ----
#pragma unroll
        for (int e = 0; e < 8; ++e) {
            pS[(half * 8 + e) * 40 + r]      = (bf16_t)s0[e];
            pS[(half * 8 + e) * 40 + r + 16] = (bf16_t)s1[e];
        }
        __syncthreads();
        const int koff = half * 8;
        const bf16_t* prow = &pS[r * 40];
        v16bf pf = load_frag16(prow + koff, prow + 16 + koff);
        __syncthreads();

        // ---- O += P(16x32) * V(32x128): V^T rows are contiguous along keys ----
#pragma unroll
        for (int t = 0; t < 8; ++t) {
            const bf16_t* vrow = VT + (size_t)(kvh * HD + t * 16 + r) * S_LEN + kb + half * 8;
            o[t] = wmma_bf16(pf, load_frag16(vrow, vrow + 16), o[t]);
        }
    }

    // finalize: divide by row sums, store bf16 [S, NH*128]
#pragma unroll
    for (int t = 0; t < 8; ++t) {
        const int col = h * HD + t * 16 + r;
#pragma unroll
        for (int e = 0; e < 8; ++e) {
            const int row = qBase + half * 8 + e;
            Oattn[(size_t)row * HID + col] = (bf16_t)(o[t][e] / rl[e]);
        }
    }
}

// ---------- host-side orchestration ----------

extern "C" void kernel_launch(void* const* d_in, const int* in_sizes, int n_in,
                              void* d_out, int out_size, void* d_ws, size_t ws_size,
                              hipStream_t stream) {
    const float* hidden = (const float*)d_in[0];
    // d_in[1] = position_ids (arange; positions derived from row index)
    const float* Wq = (const float*)d_in[2];
    const float* Wk = (const float*)d_in[3];
    const float* Wv = (const float*)d_in[4];
    const float* Wo = (const float*)d_in[5];
    float* out = (float*)d_out;

    char* ws = (char*)d_ws;
    size_t off = 0;
    auto walloc = [&](size_t bytes) -> void* {
        void* p = ws + off;
        off += (bytes + 255) & ~(size_t)255;
        return p;
    };
    bf16_t* hbf  = (bf16_t*)walloc((size_t)S_LEN * HID * 2);
    bf16_t* wqbf = (bf16_t*)walloc((size_t)HID * HID * 2);
    bf16_t* wkbf = (bf16_t*)walloc((size_t)KVD * HID * 2);
    bf16_t* wvbf = (bf16_t*)walloc((size_t)KVD * HID * 2);
    bf16_t* wobf = (bf16_t*)walloc((size_t)HID * HID * 2);
    float*  qf32 = (float*) walloc((size_t)S_LEN * HID * 4);
    float*  kf32 = (float*) walloc((size_t)S_LEN * KVD * 4);
    bf16_t* qbf  = (bf16_t*)walloc((size_t)S_LEN * HID * 2);
    bf16_t* kbf  = (bf16_t*)walloc((size_t)S_LEN * KVD * 2);
    bf16_t* vtbf = (bf16_t*)walloc((size_t)KVD * S_LEN * 2);   // V^T [d, s]
    bf16_t* abf  = (bf16_t*)walloc((size_t)S_LEN * HID * 2);
    if (off > ws_size) return;   // insufficient scratch: do nothing (deterministic)

    // 1) fp32 -> bf16 conversions (vectorized x8)
    cvt_f32_bf16_kernel<<<2048, 256, 0, stream>>>(hidden, hbf, (size_t)S_LEN * HID / 8);
    cvt_f32_bf16_kernel<<<2048, 256, 0, stream>>>(Wq, wqbf, (size_t)HID * HID / 8);
    cvt_f32_bf16_kernel<<<2048, 256, 0, stream>>>(Wk, wkbf, (size_t)KVD * HID / 8);
    cvt_f32_bf16_kernel<<<2048, 256, 0, stream>>>(Wv, wvbf, (size_t)KVD * HID / 8);
    cvt_f32_bf16_kernel<<<2048, 256, 0, stream>>>(Wo, wobf, (size_t)HID * HID / 8);

    // 2) projections (NT GEMMs, async-LDS double-buffered)
    gemm_nt_wmma_kernel<0><<<dim3(HID / 256, S_LEN / 128), 256, 0, stream>>>(
        hbf, wqbf, qf32, S_LEN, HID, HID);
    gemm_nt_wmma_kernel<0><<<dim3(KVD / 256, S_LEN / 128), 256, 0, stream>>>(
        hbf, wkbf, kf32, S_LEN, KVD, HID);
    gemm_nt_wmma_kernel<2><<<dim3(KVD / 256, S_LEN / 128), 256, 0, stream>>>(
        hbf, wvbf, vtbf, S_LEN, KVD, HID);   // stores V^T in bf16

    // 3) RoPE (fp32 -> bf16)
    rope_kernel<<<dim3(S_LEN, NH),  128, 0, stream>>>(qf32, qbf, NH);
    rope_kernel<<<dim3(S_LEN, NKV), 128, 0, stream>>>(kf32, kbf, NKV);

    // 4) causal flash attention with GQA
    flash_attn_wmma_kernel<<<dim3(S_LEN / 16, NH), 32, 0, stream>>>(qbf, kbf, vtbf, abf);

    // 5) output projection -> fp32 d_out
    gemm_nt_wmma_kernel<0><<<dim3(HID / 256, S_LEN / 128), 256, 0, stream>>>(
        abf, wobf, out, S_LEN, HID, HID);
}
